// ARPrior_34488587387584
// MI455X (gfx1250) — matrix operand order repaired and checked by reading, not compiled, not run on hardware
//
#include <hip/hip_runtime.h>

typedef __attribute__((ext_vector_type(16))) _Float16 v16h;
typedef __attribute__((ext_vector_type(8)))  float    v8f;

#define B_N   32768
#define L_N   32
#define H1_N  128
#define H2_N  64
#define ROWS_PER_BLOCK 128   // 8 waves * 16 rows each
#define THREADS 256
#define H2S_STRIDE 65        // 64 + 1 pad -> conflict-free row reads

__global__ __launch_bounds__(THREADS)
void arprior_mlp_wmma(const float* __restrict__ z,
                      const float* __restrict__ W1g,  // (L,1,128)
                      const float* __restrict__ b1g,  // (L,128)
                      const float* __restrict__ W2g,  // (L,128,64)
                      const float* __restrict__ b2g,  // (L,64)
                      const float* __restrict__ W3g,  // (L,64,2)
                      const float* __restrict__ b3g,  // (L,2)
                      float* __restrict__ out)        // [2][B][L] flat
{
    // LDS staging: W2 in f16, [n][k] so each lane's B-fragment (16 halves,
    // K contiguous) is one 32-byte aligned chunk.
    __shared__ __align__(32) _Float16 w2s[H2_N * H1_N];           // 16 KB
    __shared__ float w1s[H1_N], b1s[H1_N], b2s[H2_N];
    __shared__ float W3s[H2_N * 2], b3s[2], xs[ROWS_PER_BLOCK];
    __shared__ float h2s[8 * 16 * H2S_STRIDE];                    // 33.3 KB

    const int tid = threadIdx.x;
    const int l   = blockIdx.y;                  // latent index, uniform per block
    const int b0  = blockIdx.x * ROWS_PER_BLOCK; // first batch row of this block

    // ---- stage per-l parameters ----
    for (int i = tid; i < H1_N; i += THREADS) {
        w1s[i] = W1g[l * H1_N + i];
        b1s[i] = b1g[l * H1_N + i];
    }
    for (int i = tid; i < H2_N; i += THREADS) b2s[i] = b2g[l * H2_N + i];
    for (int i = tid; i < H2_N * 2; i += THREADS) W3s[i] = W3g[l * H2_N * 2 + i];
    if (tid < 2) b3s[tid] = b3g[l * 2 + tid];

    // W2[l] : (128,64) row-major in global; store f16 transposed [n][k] in LDS.
    const float* W2l = W2g + (size_t)l * H1_N * H2_N;
    for (int e = tid; e < H1_N * H2_N; e += THREADS) {      // coalesced global reads
        int k = e >> 6, n = e & 63;
        w2s[n * H1_N + k] = (_Float16)W2l[e];
    }

    // per-row scalar input x = (l==0) ? 0 : mean(z[b,0:l])
    if (tid < ROWS_PER_BLOCK) {
        const float* zr = z + (size_t)(b0 + tid) * L_N;
        float s = 0.f;
        for (int j = 0; j < l; ++j) s += zr[j];
        xs[tid] = (l == 0) ? 0.f : s * (1.0f / (float)l);
    }
    __syncthreads();

    const int w  = tid >> 5;        // wave id 0..7
    const int ln = tid & 31;        // lane
    const int lo = ln & 15;
    const int hi = ln >> 4;

    // ---- build A fragments: h1 = relu(x*w1 + b1), f16, 16x32 A, contiguous K = hi*16 + s ----
    // lane (lo,hi): row M = lo; VGPR p -> K = hi*16 + 2p + half
    const float x = xs[w * 16 + lo];
    v16h afrag[4];
#pragma unroll
    for (int q = 0; q < 4; ++q) {
#pragma unroll
        for (int p = 0; p < 8; ++p) {
            const int kb = q * 32 + 2 * p + hi * 16;
            float h0 = fmaf(x, w1s[kb],     b1s[kb]);     h0 = h0 > 0.f ? h0 : 0.f;
            float h1 = fmaf(x, w1s[kb + 1], b1s[kb + 1]); h1 = h1 > 0.f ? h1 : 0.f;
            afrag[q][2 * p]     = (_Float16)h0;
            afrag[q][2 * p + 1] = (_Float16)h1;
        }
    }

    // ---- layer 2 GEMM: 16x64 = (16x128)x(128x64), 4 N-tiles x 4 K-steps ----
    const int n_col = lo;                // B/C column within tile
#pragma unroll
    for (int t = 0; t < 4; ++t) {
        v8f acc = {};
#pragma unroll
        for (int q = 0; q < 4; ++q) {
            // B fragment: lane column n = t*16+lo, K range = q*32 + hi*16 .. +15
            const int n    = t * 16 + n_col;
            const int koff = q * 32 + hi * 16;
            const v16h bfrag = *(const v16h*)(&w2s[n * H1_N + koff]);
            acc = __builtin_amdgcn_wmma_f32_16x16x32_f16(
                      false, afrag[q], false, bfrag, (short)0, acc, false, false);
        }
        // epilogue: bias + ReLU -> LDS h2 tile (C layout: VGPR r -> row r+hi*8)
#pragma unroll
        for (int r = 0; r < 8; ++r) {
            const int row = r + hi * 8;
            const int n   = t * 16 + n_col;
            float v = acc[r] + b2s[n];
            h2s[(w * 16 + row) * H2S_STRIDE + n] = v > 0.f ? v : 0.f;
        }
    }
    __syncthreads();

    // ---- layer 3: 64 -> 2. wave32 = 16 rows x 2 output cols, one per lane ----
    const int row = ln >> 1;
    const int c   = ln & 1;
    const float* hrow = &h2s[(w * 16 + row) * H2S_STRIDE];
    float sum = b3s[c];
#pragma unroll 8
    for (int n = 0; n < H2_N; ++n) sum = fmaf(hrow[n], W3s[n * 2 + c], sum);

    const int bg = b0 + w * 16 + row;
    out[(size_t)c * B_N * L_N + (size_t)bg * L_N + l] = sum;
}

extern "C" void kernel_launch(void* const* d_in, const int* in_sizes, int n_in,
                              void* d_out, int out_size, void* d_ws, size_t ws_size,
                              hipStream_t stream) {
    (void)in_sizes; (void)n_in; (void)d_ws; (void)ws_size; (void)out_size;
    const float* z  = (const float*)d_in[0];
    const float* W1 = (const float*)d_in[1];
    const float* b1 = (const float*)d_in[2];
    const float* W2 = (const float*)d_in[3];
    const float* b2 = (const float*)d_in[4];
    const float* W3 = (const float*)d_in[5];
    const float* b3 = (const float*)d_in[6];
    float* out = (float*)d_out;

    dim3 grid(B_N / ROWS_PER_BLOCK, L_N, 1);   // (256, 32)
    dim3 block(THREADS, 1, 1);
    arprior_mlp_wmma<<<grid, block, 0, stream>>>(z, W1, b1, W2, b2, W3, b3, out);
}
